// PointTransformerBlock_54786602828253
// MI455X (gfx1250) — matrix operand order repaired and checked by reading, not compiled, not run on hardware
//
#include <hip/hip_runtime.h>
#include <hip/hip_bf16.h>

typedef __attribute__((ext_vector_type(16))) _Float16 v16h;
typedef __attribute__((ext_vector_type(8)))  float    v8f;
typedef unsigned int u32x4 __attribute__((ext_vector_type(4)));
typedef int          i32x4 __attribute__((ext_vector_type(4)));
typedef int          i32x8 __attribute__((ext_vector_type(8)));

#define B_    2
#define N_    16384
#define K_    16
#define C_    128
#define ATTN_ 16
#define BN_   (B_ * N_)        // 32768
#define BNK_  (BN_ * K_)       // 524288
#define BIG_  ((size_t)BN_ * C_)
#define EPS_  1e-5f

// f16 weight region layout (halves), after stats
#define WH_W1   0
#define WH_WQ   16384
#define WH_WK   32768
#define WH_WV   49152
#define WH_W3   65536
#define WH_G1   81920        // fg_w1 col-major: [a*128 + k], 2048 halves
#define WH_G2   83968        // fg_w2 col-major K-padded to 32: [n*32 + k], 512 halves
#define WH_TOT  84480

// ---------- WMMA helpers (CDNA5 layouts, CDNA5 ISA §7.12.2) ----------

__device__ __forceinline__ v8f wmma_f16(v16h a, v16h b, v8f c) {
    return __builtin_amdgcn_wmma_f32_16x16x32_f16(false, a, false, b, (short)0, c,
                                                  false, false);
}

// 8 contiguous floats (2 x b128 / ds_load_b128)
__device__ __forceinline__ void ld8(const float* __restrict__ p, int k0, float* dst) {
    float4 a = *(const float4*)(p + k0);
    float4 b = *(const float4*)(p + k0 + 4);
    dst[0] = a.x; dst[1] = a.y; dst[2] = a.z; dst[3] = a.w;
    dst[4] = b.x; dst[5] = b.y; dst[6] = b.z; dst[7] = b.w;
}

__device__ __forceinline__ float4 ld4(const float* __restrict__ p) {
    return *(const float4*)p;
}

// ---------- TDM staging: async-load a 16x128 fp32 tile (8KB) into LDS ----------
// D# built per CDNA5 ISA: 2D tensor, data_size=4B,
// tile_dim0=128, tile_dim1=16, tensor_dim0_stride=128, type=2, count=1.

__device__ __forceinline__ void stage_tile(const float* __restrict__ gsrc,
                                           float* lbuf, int tid, int wave) {
#if __has_builtin(__builtin_amdgcn_tensor_load_to_lds)
    if (wave == 0) {
        unsigned long long ga = (unsigned long long)(uintptr_t)gsrc;
        unsigned int lds = (unsigned int)(uintptr_t)lbuf;   // LDS aperture: addr[31:0]
        u32x4 g0;
        g0.x = 1u;                                          // count=1 (valid user D#)
        g0.y = lds;                                         // lds_addr [63:32]
        g0.z = (unsigned int)ga;                            // global_addr [95:64]
        g0.w = (unsigned int)((ga >> 32) & 0x1FFFFFFu) | (2u << 30);  // [120:96]+type=2
        i32x8 g1;
        g1[0] = (int)(2u << 16);      // workgroup_mask=0, data_size=2 (4B)
        g1[1] = (int)(128u << 16);    // tensor_dim0[15:0]=128 in bits [79:64]... [63:48]
        g1[2] = (int)(16u << 16);     // tensor_dim0 hi=0; tensor_dim1[15:0]=16
        g1[3] = (int)(128u << 16);    // tensor_dim1 hi=0; tile_dim0=128
        g1[4] = 16;                   // tile_dim1=16 (tile_dim2=0)
        g1[5] = 128;                  // tensor_dim0_stride[31:0]=128
        g1[6] = 0;                    // stride hi, dim1_stride lo
        g1[7] = 0;                    // dim1_stride hi
        i32x4 z4 = {0, 0, 0, 0};
#if defined(__clang_major__) && (__clang_major__ >= 23)
        i32x8 z8 = {0, 0, 0, 0, 0, 0, 0, 0};
        __builtin_amdgcn_tensor_load_to_lds(g0, g1, z4, z4, z8, 0);
#else
        __builtin_amdgcn_tensor_load_to_lds(g0, g1, z4, z4, 0);
#endif
        __builtin_amdgcn_s_wait_tensorcnt(0);
    }
#else
    for (int i = tid; i < 16 * C_ / 4; i += 256)
        ((float4*)lbuf)[i] = ((const float4*)gsrc)[i];
#endif
}

// ---------- small utility kernels ----------

__global__ void k_zero(float* p, int n) {
    for (int i = threadIdx.x; i < n; i += blockDim.x) p[i] = 0.f;
}

// Convert weights to f16 column-major for contiguous B-fragment loads.
__global__ void k_prep_w(const float* __restrict__ w1, const float* __restrict__ wq,
                         const float* __restrict__ wk, const float* __restrict__ wv,
                         const float* __restrict__ w3, const float* __restrict__ g1,
                         const float* __restrict__ g2, _Float16* __restrict__ out) {
    int i = blockIdx.x * blockDim.x + threadIdx.x;
    if (i < 5 * 16384) {
        int mtx = i >> 14, r = i & 16383;
        int n = r >> 7, k = r & 127;
        const float* W = (mtx == 0) ? w1 : (mtx == 1) ? wq : (mtx == 2) ? wk
                       : (mtx == 3) ? wv : w3;
        out[i] = (_Float16)W[k * C_ + n];
    } else if (i < WH_G2) {
        int r = i - WH_G1;
        int a = r >> 7, k = r & 127;
        out[i] = (_Float16)g1[k * ATTN_ + a];
    } else if (i < WH_TOT) {
        int r = i - WH_G2;
        int n = r >> 5, k = r & 31;
        out[i] = (_Float16)((k < ATTN_) ? g2[k * ATTN_ + n] : 0.f);
    }
}

// scale = g * rsqrt(var+eps); shift = b - mean*scale
__global__ void k_finalize(const float* __restrict__ sum, const float* __restrict__ sq,
                           const float* __restrict__ g,   const float* __restrict__ b,
                           float* __restrict__ sc, float* __restrict__ sh,
                           int nch, float count) {
    int i = blockIdx.x * blockDim.x + threadIdx.x;
    if (i < nch) {
        float m = sum[i] / count;
        float v = sq[i] / count - m * m;
        float s = g[i] * rsqrtf(v + EPS_);
        sc[i] = s;
        sh[i] = b[i] - m * s;
    }
}

// ---------- WMMA GEMM (BNx128)@(128x128), optional input affine+relu, + BN stats ----------

__global__ void k_gemm_bn(const float* __restrict__ Xin, const _Float16* __restrict__ Wt,
                          const float* __restrict__ sc,  const float* __restrict__ sh,
                          int do_act, float* __restrict__ Xout,
                          float* __restrict__ gsum, float* __restrict__ gsq) {
    __shared__ float lbuf[16 * C_];          // A tile, then D tile
    __shared__ float ssum[C_], ssq[C_];
    int tid = threadIdx.x, wave = tid >> 5, lane = tid & 31;
    if (tid < C_) { ssum[tid] = 0.f; ssq[tid] = 0.f; }

    int row0 = blockIdx.x * 16;
    int col0 = wave * 16;
    int m = lane & 15;
    int alane = (lane >> 4) << 3;               // D row base: 0 or 8; A K-base = 2*alane: 0 or 16

    stage_tile(Xin + (size_t)row0 * C_, lbuf, tid, wave);
    __syncthreads();

    const float* arow = &lbuf[m * C_];
    v16h afrag[4];
#pragma unroll
    for (int kc = 0; kc < 4; kc++) {
        int k0 = kc * 32 + 2 * alane;
        float x[16];
        ld8(arow, k0, x);
        ld8(arow, k0 + 8, x + 8);
        if (do_act) {
            float s8[16], h8[16];
            ld8(sc, k0, s8); ld8(sc, k0 + 8, s8 + 8);
            ld8(sh, k0, h8); ld8(sh, k0 + 8, h8 + 8);
#pragma unroll
            for (int h = 0; h < 16; h++) {
                float v = x[h] * s8[h] + h8[h];
                x[h] = v > 0.f ? v : 0.f;
            }
        }
#pragma unroll
        for (int h = 0; h < 16; h++) afrag[kc][h] = (_Float16)x[h];
    }
    __syncthreads();                             // all A reads done before D writes

    v8f acc = {};
    int colB = col0 + (lane & 15);
    int kb = (lane >> 4) << 4;                  // B K-base: 0 or 16
#pragma unroll
    for (int kc = 0; kc < 4; kc++) {
        v16h bf = *(const v16h*)(Wt + colB * C_ + kc * 32 + kb);
        acc = wmma_f16(afrag[kc], bf, acc);
    }

#pragma unroll
    for (int r = 0; r < 8; r++) lbuf[(alane + r) * C_ + colB] = acc[r];
    __syncthreads();

    float* dst = Xout + (size_t)row0 * C_;
    for (int i = tid; i < 16 * C_ / 4; i += 256) {
        float4 v = ((const float4*)lbuf)[i];
        ((float4*)dst)[i] = v;                   // coalesced b128 store
        int c0 = (i * 4) & (C_ - 1);
        atomicAdd(&ssum[c0 + 0], v.x); atomicAdd(&ssq[c0 + 0], v.x * v.x);
        atomicAdd(&ssum[c0 + 1], v.y); atomicAdd(&ssq[c0 + 1], v.y * v.y);
        atomicAdd(&ssum[c0 + 2], v.z); atomicAdd(&ssq[c0 + 2], v.z * v.z);
        atomicAdd(&ssum[c0 + 3], v.w); atomicAdd(&ssq[c0 + 3], v.w * v.w);
    }
    __syncthreads();
    if (tid < C_) { atomicAdd(&gsum[tid], ssum[tid]); atomicAdd(&gsq[tid], ssq[tid]); }
}

// ---------- Q/K/V projections: A = relu(bn1(X1)); 12 WMMAs per wave ----------

__global__ void k_qkv(const float* __restrict__ X1, const float* __restrict__ sc,
                      const float* __restrict__ sh,
                      const _Float16* __restrict__ WQt, const float* __restrict__ bq,
                      const _Float16* __restrict__ WKt, const float* __restrict__ bk,
                      const _Float16* __restrict__ WVt, const float* __restrict__ bv,
                      float* __restrict__ Q, float* __restrict__ Kf, float* __restrict__ Vf) {
    __shared__ float lbuf[16 * C_];
    int tid = threadIdx.x, wave = tid >> 5, lane = tid & 31;
    int row0 = blockIdx.x * 16;
    int col0 = wave * 16;
    int m = lane & 15;
    int alane = (lane >> 4) << 3;

    stage_tile(X1 + (size_t)row0 * C_, lbuf, tid, wave);
    __syncthreads();

    const float* arow = &lbuf[m * C_];
    v16h afrag[4];
#pragma unroll
    for (int kc = 0; kc < 4; kc++) {
        int k0 = kc * 32 + 2 * alane;
        float x[16], s8[16], h8[16];
        ld8(arow, k0, x);      ld8(arow, k0 + 8, x + 8);
        ld8(sc, k0, s8);       ld8(sc, k0 + 8, s8 + 8);
        ld8(sh, k0, h8);       ld8(sh, k0 + 8, h8 + 8);
#pragma unroll
        for (int h = 0; h < 16; h++) {
            float v = x[h] * s8[h] + h8[h];
            afrag[kc][h] = (_Float16)(v > 0.f ? v : 0.f);
        }
    }
    __syncthreads();

    v8f aq = {}, ak = {}, av = {};
    int colB = col0 + (lane & 15);
    int kb = (lane >> 4) << 4;
#pragma unroll
    for (int kc = 0; kc < 4; kc++) {
        int wo = colB * C_ + kc * 32 + kb;
        v16h bfq = *(const v16h*)(WQt + wo);
        v16h bfk = *(const v16h*)(WKt + wo);
        v16h bfv = *(const v16h*)(WVt + wo);
        aq = wmma_f16(afrag[kc], bfq, aq);
        ak = wmma_f16(afrag[kc], bfk, ak);
        av = wmma_f16(afrag[kc], bfv, av);
    }

    float bqv = bq[colB], bkv = bk[colB], bvv = bv[colB];
#pragma unroll
    for (int which = 0; which < 3; which++) {
        float bias = (which == 0) ? bqv : (which == 1) ? bkv : bvv;
        float* dst = ((which == 0) ? Q : (which == 1) ? Kf : Vf) + (size_t)row0 * C_;
#pragma unroll
        for (int r = 0; r < 8; r++) {
            float v = (which == 0) ? aq[r] : (which == 1) ? ak[r] : av[r];
            lbuf[(alane + r) * C_ + colB] = v + bias;
        }
        __syncthreads();
        for (int i = tid; i < 16 * C_ / 4; i += 256)
            ((float4*)dst)[i] = ((const float4*)lbuf)[i];
        __syncthreads();
    }
}

// ---------- fd BN stats over rel@fd_w1+fd_b1 (3 channels, BNK samples) ----------

__global__ void k_fd_stats(const float* __restrict__ xyz, const int* __restrict__ idx,
                           const float* __restrict__ fdw1, const float* __restrict__ fdb1,
                           float* __restrict__ gsum, float* __restrict__ gsq) {
    __shared__ float ssum[4], ssq[4];
    int tid = threadIdx.x;
    if (tid < 4) { ssum[tid] = 0.f; ssq[tid] = 0.f; }
    __syncthreads();
    int i = blockIdx.x * blockDim.x + tid;   // over BNK_
    int bn = i >> 4;
    int b  = bn / N_;
    int j  = idx[i];
    const float* p  = xyz + (size_t)bn * 3;
    const float* pn = xyz + (size_t)(b * N_ + j) * 3;
    float r0 = p[0] - pn[0], r1 = p[1] - pn[1], r2 = p[2] - pn[2];
#pragma unroll
    for (int c = 0; c < 3; c++) {
        float t = r0 * fdw1[c] + r1 * fdw1[3 + c] + r2 * fdw1[6 + c] + fdb1[c];
        atomicAdd(&ssum[c], t);
        atomicAdd(&ssq[c], t * t);
    }
    __syncthreads();
    if (tid < 3) { atomicAdd(&gsum[tid], ssum[tid]); atomicAdd(&gsq[tid], ssq[tid]); }
}

// ---------- per-point neighbor prep (post-activation t3, idx cache) ----------

__device__ __forceinline__ void prep_point(int wave, int lane, int pt, int b,
                                           const int* __restrict__ idx,
                                           const float* __restrict__ xyz,
                                           const float* __restrict__ fdw1,
                                           const float* __restrict__ fdb1,
                                           const float* __restrict__ fdsc,
                                           const float* __restrict__ fdsh,
                                           float (*t3)[K_][3], int (*sidx)[K_]) {
    if (lane < K_) {
        int j = idx[(size_t)pt * K_ + lane];
        sidx[wave][lane] = j;
        const float* p  = xyz + (size_t)pt * 3;
        const float* pn = xyz + (size_t)(b * N_ + j) * 3;
        float r0 = p[0] - pn[0], r1 = p[1] - pn[1], r2 = p[2] - pn[2];
#pragma unroll
        for (int c = 0; c < 3; c++) {
            float t = r0 * fdw1[c] + r1 * fdw1[3 + c] + r2 * fdw1[6 + c] + fdb1[c];
            t = t * fdsc[c] + fdsh[c];
            t3[wave][lane][c] = t > 0.f ? t : 0.f;
        }
    }
}

// ---------- attn0 stats: per-channel stats of q - k + pos_enc over BNK ----------

__global__ void k_attn0_stats(const float* __restrict__ Q, const float* __restrict__ Kf,
                              const int* __restrict__ idx, const float* __restrict__ xyz,
                              const float* __restrict__ fdw1, const float* __restrict__ fdb1,
                              const float* __restrict__ fdsc, const float* __restrict__ fdsh,
                              const float* __restrict__ fdw2, const float* __restrict__ fdb2,
                              float* __restrict__ gsum, float* __restrict__ gsq) {
    __shared__ float t3[8][K_][3];
    __shared__ int   sidx[8][K_];
    __shared__ float ssum[C_], ssq[C_];
    int tid = threadIdx.x, wave = tid >> 5, lane = tid & 31;
    if (tid < C_) { ssum[tid] = 0.f; ssq[tid] = 0.f; }
    int pt = blockIdx.x * 8 + wave;
    int b  = pt / N_;
    prep_point(wave, lane, pt, b, idx, xyz, fdw1, fdb1, fdsc, fdsh, t3, sidx);
    __syncthreads();

    int c0 = lane * 4;
    float4 qv = ld4(Q + (size_t)pt * C_ + c0);
    float4 u0 = ld4(fdw2 + c0), u1 = ld4(fdw2 + C_ + c0);
    float4 u2 = ld4(fdw2 + 2 * C_ + c0), ub = ld4(fdb2 + c0);
    float s[4] = {0, 0, 0, 0}, s2[4] = {0, 0, 0, 0};

    for (int m = 0; m < K_; m++) {
        int j = sidx[wave][m];
        float4 kv = ld4(Kf + (size_t)(b * N_ + j) * C_ + c0);
        float p0 = t3[wave][m][0], p1 = t3[wave][m][1], p2 = t3[wave][m][2];
        float a0 = qv.x - kv.x + p0 * u0.x + p1 * u1.x + p2 * u2.x + ub.x;
        float a1 = qv.y - kv.y + p0 * u0.y + p1 * u1.y + p2 * u2.y + ub.y;
        float a2 = qv.z - kv.z + p0 * u0.z + p1 * u1.z + p2 * u2.z + ub.z;
        float a3 = qv.w - kv.w + p0 * u0.w + p1 * u1.w + p2 * u2.w + ub.w;
        s[0] += a0; s2[0] += a0 * a0;
        s[1] += a1; s2[1] += a1 * a1;
        s[2] += a2; s2[2] += a2 * a2;
        s[3] += a3; s2[3] += a3 * a3;
    }
#pragma unroll
    for (int cc = 0; cc < 4; cc++) {
        atomicAdd(&ssum[c0 + cc], s[cc]);
        atomicAdd(&ssq[c0 + cc], s2[cc]);
    }
    __syncthreads();
    if (tid < C_) { atomicAdd(&gsum[tid], ssum[tid]); atomicAdd(&gsq[tid], ssq[tid]); }
}

// ---------- per-point attn1 = relu(bn(attn0)) @ fg_w1 via 4 WMMAs (K=128) ----------

__device__ __forceinline__ v8f compute_attn1(const float* __restrict__ qrow,
                                             const float* __restrict__ krow,
                                             float p0, float p1, float p2,
                                             const float* __restrict__ fdw2,
                                             const float* __restrict__ fdb2,
                                             const float* __restrict__ fg1sc,
                                             const float* __restrict__ fg1sh,
                                             const _Float16* __restrict__ Wg1t, int lane) {
    v8f acc = {};
    int alane = (lane >> 4) << 3;
    int nB = lane & 15, kb = (lane >> 4) << 4;
#pragma unroll
    for (int kc = 0; kc < 4; kc++) {
        v16h af;
#pragma unroll
        for (int g = 0; g < 2; g++) {
            int k0 = kc * 32 + 2 * alane + g * 8;
            float q8[8], k8[8], w0[8], w1[8], w2[8], b8[8], c8[8], h8[8];
            ld8(qrow, k0, q8);        ld8(krow, k0, k8);
            ld8(fdw2, k0, w0);        ld8(fdw2 + C_, k0, w1);
            ld8(fdw2 + 2 * C_, k0, w2); ld8(fdb2, k0, b8);
            ld8(fg1sc, k0, c8);       ld8(fg1sh, k0, h8);
#pragma unroll
            for (int h = 0; h < 8; h++) {
                float pe = p0 * w0[h] + p1 * w1[h] + p2 * w2[h] + b8[h];
                float a0 = (q8[h] - k8[h] + pe) * c8[h] + h8[h];
                af[g * 8 + h] = (_Float16)(a0 > 0.f ? a0 : 0.f);
            }
        }
        v16h bf = *(const v16h*)(Wg1t + nB * C_ + kc * 32 + kb);
        acc = wmma_f16(af, bf, acc);
    }
    return acc;
}

// ---------- attn1 stats: per-ATTN-channel stats over BNK ----------

__global__ void k_attn1_stats(const float* __restrict__ Q, const float* __restrict__ Kf,
                              const int* __restrict__ idx, const float* __restrict__ xyz,
                              const float* __restrict__ fdw1, const float* __restrict__ fdb1,
                              const float* __restrict__ fdsc, const float* __restrict__ fdsh,
                              const float* __restrict__ fdw2, const float* __restrict__ fdb2,
                              const float* __restrict__ fg1sc, const float* __restrict__ fg1sh,
                              const _Float16* __restrict__ Wg1t, const float* __restrict__ fgb1,
                              float* __restrict__ gsum, float* __restrict__ gsq) {
    __shared__ float t3[8][K_][3];
    __shared__ int   sidx[8][K_];
    __shared__ float ssum[ATTN_], ssq[ATTN_];
    int tid = threadIdx.x, wave = tid >> 5, lane = tid & 31;
    if (tid < ATTN_) { ssum[tid] = 0.f; ssq[tid] = 0.f; }
    int pt = blockIdx.x * 8 + wave;
    int b  = pt / N_;
    prep_point(wave, lane, pt, b, idx, xyz, fdw1, fdb1, fdsc, fdsh, t3, sidx);
    __syncthreads();

    int m = lane & 15;
    int j = sidx[wave][m];
    v8f a1 = compute_attn1(Q + (size_t)pt * C_, Kf + (size_t)(b * N_ + j) * C_,
                           t3[wave][m][0], t3[wave][m][1], t3[wave][m][2],
                           fdw2, fdb2, fg1sc, fg1sh, Wg1t, lane);
    int a = lane & 15;
    float s = 0.f, s2 = 0.f;
#pragma unroll
    for (int r = 0; r < 8; r++) {
        float v = a1[r] + fgb1[a];
        s += v; s2 += v * v;
    }
    atomicAdd(&ssum[a], s);
    atomicAdd(&ssq[a], s2);
    __syncthreads();
    if (tid < ATTN_) { atomicAdd(&gsum[tid], ssum[tid]); atomicAdd(&gsq[tid], ssq[tid]); }
}

// ---------- main pass: attn2, softmax over K, aggregation, bn2 stats ----------

__global__ void k_main(const float* __restrict__ Q, const float* __restrict__ Kf,
                       const float* __restrict__ Vf, const int* __restrict__ idx,
                       const float* __restrict__ xyz,
                       const float* __restrict__ fdw1, const float* __restrict__ fdb1,
                       const float* __restrict__ fdsc, const float* __restrict__ fdsh,
                       const float* __restrict__ fdw2, const float* __restrict__ fdb2,
                       const float* __restrict__ fg1sc, const float* __restrict__ fg1sh,
                       const _Float16* __restrict__ Wg1t, const float* __restrict__ fgb1,
                       const float* __restrict__ fg2sc, const float* __restrict__ fg2sh,
                       const _Float16* __restrict__ Wg2t, const float* __restrict__ fgb2,
                       float* __restrict__ RES,
                       float* __restrict__ gsum, float* __restrict__ gsq) {
    __shared__ float t3[8][K_][3];
    __shared__ int   sidx[8][K_];
    __shared__ float sA[8][K_][ATTN_];   // relu(bn2(attn1)) staging (D -> A layout)
    __shared__ float sW[8][K_][ATTN_];   // softmax weights
    __shared__ float ssum[C_], ssq[C_];
    int tid = threadIdx.x, wave = tid >> 5, lane = tid & 31;
    if (tid < C_) { ssum[tid] = 0.f; ssq[tid] = 0.f; }
    int pt = blockIdx.x * 8 + wave;
    int b  = pt / N_;
    prep_point(wave, lane, pt, b, idx, xyz, fdw1, fdb1, fdsc, fdsh, t3, sidx);
    __syncthreads();

    int m = lane & 15;
    int alane = (lane >> 4) << 3;
    int j = sidx[wave][m];
    v8f a1 = compute_attn1(Q + (size_t)pt * C_, Kf + (size_t)(b * N_ + j) * C_,
                           t3[wave][m][0], t3[wave][m][1], t3[wave][m][2],
                           fdw2, fdb2, fg1sc, fg1sh, Wg1t, lane);

    int aCol = lane & 15;
#pragma unroll
    for (int r = 0; r < 8; r++) {
        float v = a1[r] + fgb1[aCol];
        v = v * fg2sc[aCol] + fg2sh[aCol];
        sA[wave][alane + r][aCol] = v > 0.f ? v : 0.f;
    }
    __syncthreads();

    // attn2 = sA(16x16) @ fg_w2(16x16), K zero-padded to 32 -> one WMMA.
    v16h af;
#pragma unroll
    for (int h = 0; h < 16; h++) af[h] = (lane < 16) ? (_Float16)sA[wave][m][h] : (_Float16)0.f;
    int kb = (lane >> 4) << 4;
    v16h bf = *(const v16h*)(Wg2t + aCol * 32 + kb);   // zero-padded K>=16
    v8f zero = {};
    v8f a2 = wmma_f16(af, bf, zero);

    // softmax over neighbor dim: lane n and lane n+16 share column a=n
    float vals[8];
    float mx = -1e30f;
#pragma unroll
    for (int r = 0; r < 8; r++) { vals[r] = a2[r] + fgb2[aCol]; mx = fmaxf(mx, vals[r]); }
    mx = fmaxf(mx, __shfl_xor(mx, 16, 32));
    float s = 0.f;
#pragma unroll
    for (int r = 0; r < 8; r++) { vals[r] = __expf(vals[r] - mx); s += vals[r]; }
    s += __shfl_xor(s, 16, 32);
    float inv = 1.f / s;
#pragma unroll
    for (int r = 0; r < 8; r++) sW[wave][alane + r][aCol] = vals[r] * inv;
    __syncthreads();

    // aggregation: res[c] = sum_m (Vf[idx[m]][c] + pos_enc[m][c]) * w[m][c%16]
    int c0 = lane * 4;
    int a0i = c0 & (ATTN_ - 1);
    float4 u0 = ld4(fdw2 + c0), u1 = ld4(fdw2 + C_ + c0);
    float4 u2 = ld4(fdw2 + 2 * C_ + c0), ub = ld4(fdb2 + c0);
    float acc[4] = {0.f, 0.f, 0.f, 0.f};
    for (int mm = 0; mm < K_; mm++) {
        int jj = sidx[wave][mm];
        float4 vv = ld4(Vf + (size_t)(b * N_ + jj) * C_ + c0);
        float p0 = t3[wave][mm][0], p1 = t3[wave][mm][1], p2 = t3[wave][mm][2];
        acc[0] += (vv.x + p0 * u0.x + p1 * u1.x + p2 * u2.x + ub.x) * sW[wave][mm][a0i + 0];
        acc[1] += (vv.y + p0 * u0.y + p1 * u1.y + p2 * u2.y + ub.y) * sW[wave][mm][a0i + 1];
        acc[2] += (vv.z + p0 * u0.z + p1 * u1.z + p2 * u2.z + ub.z) * sW[wave][mm][a0i + 2];
        acc[3] += (vv.w + p0 * u0.w + p1 * u1.w + p2 * u2.w + ub.w) * sW[wave][mm][a0i + 3];
    }
    float4 o; o.x = acc[0]; o.y = acc[1]; o.z = acc[2]; o.w = acc[3];
    *(float4*)(RES + (size_t)pt * C_ + c0) = o;
#pragma unroll
    for (int cc = 0; cc < 4; cc++) {
        atomicAdd(&ssum[c0 + cc], acc[cc]);
        atomicAdd(&ssq[c0 + cc], acc[cc] * acc[cc]);
    }
    __syncthreads();
    if (tid < C_) { atomicAdd(&gsum[tid], ssum[tid]); atomicAdd(&gsq[tid], ssq[tid]); }
}

// ---------- epilogue: out = relu(bn3(X3) + identity); prepend xyz ----------

__global__ void k_out(const float* __restrict__ X3, const float* __restrict__ sc,
                      const float* __restrict__ sh, const float* __restrict__ feat,
                      const float* __restrict__ xyz, float* __restrict__ out) {
    size_t i = (size_t)blockIdx.x * blockDim.x + threadIdx.x;  // over BIG_/4
    if (i < (size_t)BN_ * 3 / 4)
        *((float4*)out + i) = *((const float4*)xyz + i);
    if (i < BIG_ / 4) {
        int c0 = (int)((i * 4) & (C_ - 1));
        float4 x = *((const float4*)X3 + i);
        float4 f = *((const float4*)feat + i);
        float4 o;
        o.x = x.x * sc[c0 + 0] + sh[c0 + 0] + f.x;
        o.y = x.y * sc[c0 + 1] + sh[c0 + 1] + f.y;
        o.z = x.z * sc[c0 + 2] + sh[c0 + 2] + f.z;
        o.w = x.w * sc[c0 + 3] + sh[c0 + 3] + f.w;
        o.x = o.x > 0.f ? o.x : 0.f;
        o.y = o.y > 0.f ? o.y : 0.f;
        o.z = o.z > 0.f ? o.z : 0.f;
        o.w = o.w > 0.f ? o.w : 0.f;
        *(float4*)(out + (size_t)BN_ * 3 + i * 4) = o;
    }
}

// ---------- launch ----------

extern "C" void kernel_launch(void* const* d_in, const int* in_sizes, int n_in,
                              void* d_out, int out_size, void* d_ws, size_t ws_size,
                              hipStream_t stream) {
    (void)in_sizes; (void)n_in; (void)out_size; (void)ws_size;
    const float* xyz      = (const float*)d_in[0];
    const float* feat     = (const float*)d_in[1];
    const int*   knn      = (const int*)  d_in[2];
    const float* w1       = (const float*)d_in[3];
    const float* bn1_g    = (const float*)d_in[4];
    const float* bn1_b    = (const float*)d_in[5];
    const float* lq_w     = (const float*)d_in[6];
    const float* lq_b     = (const float*)d_in[7];
    const float* lk_w     = (const float*)d_in[8];
    const float* lk_b     = (const float*)d_in[9];
    const float* lv_w     = (const float*)d_in[10];
    const float* lv_b     = (const float*)d_in[11];
    const float* fd_w1    = (const float*)d_in[12];
    const float* fd_b1    = (const float*)d_in[13];
    const float* fd_bn_g  = (const float*)d_in[14];
    const float* fd_bn_b  = (const float*)d_in[15];
    const float* fd_w2    = (const float*)d_in[16];
    const float* fd_b2    = (const float*)d_in[17];
    const float* fg_bn1_g = (const float*)d_in[18];
    const float* fg_bn1_b = (const float*)d_in[19];
    const float* fg_w1    = (const float*)d_in[20];
    const float* fg_b1    = (const float*)d_in[21];
    const float* fg_bn2_g = (const float*)d_in[22];
    const float* fg_bn2_b = (const float*)d_in[23];
    const float* fg_w2    = (const float*)d_in[24];
    const float* fg_b2    = (const float*)d_in[25];
    const float* bn2_g    = (const float*)d_in[26];
    const float* bn2_b    = (const float*)d_in[27];
    const float* w3       = (const float*)d_in[28];
    const float* bn3_g    = (const float*)d_in[29];
    const float* bn3_b    = (const float*)d_in[30];

    float* ws = (float*)d_ws;
    float* X1 = ws;              // reused as RES after k_qkv
    float* Qb = ws + BIG_;       // reused as X3 after k_main
    float* Kb = ws + 2 * BIG_;
    float* Vb = ws + 3 * BIG_;
    float* st = ws + 4 * BIG_;   // stats: 24 slots x 128 floats
    _Float16* wh = (_Float16*)(st + 24 * 128);   // f16 weights, 32B-aligned
#define SLOT(i) (st + (i) * 128)

    k_zero<<<1, 256, 0, stream>>>(st, 12 * 128);
    k_prep_w<<<(WH_TOT + 255) / 256, 256, 0, stream>>>(w1, lq_w, lk_w, lv_w, w3,
                                                       fg_w1, fg_w2, wh);

    k_gemm_bn<<<BN_ / 16, 256, 0, stream>>>(feat, wh + WH_W1, nullptr, nullptr, 0, X1,
                                            SLOT(0), SLOT(1));
    k_finalize<<<1, 128, 0, stream>>>(SLOT(0), SLOT(1), bn1_g, bn1_b,
                                      SLOT(12), SLOT(13), C_, (float)BN_);

    k_qkv<<<BN_ / 16, 256, 0, stream>>>(X1, SLOT(12), SLOT(13),
                                        wh + WH_WQ, lq_b, wh + WH_WK, lk_b,
                                        wh + WH_WV, lv_b, Qb, Kb, Vb);

    k_fd_stats<<<BNK_ / 256, 256, 0, stream>>>(xyz, knn, fd_w1, fd_b1, SLOT(2), SLOT(3));
    k_finalize<<<1, 128, 0, stream>>>(SLOT(2), SLOT(3), fd_bn_g, fd_bn_b,
                                      SLOT(14), SLOT(15), 3, (float)BNK_);

    k_attn0_stats<<<BN_ / 8, 256, 0, stream>>>(Qb, Kb, knn, xyz, fd_w1, fd_b1,
                                               SLOT(14), SLOT(15), fd_w2, fd_b2,
                                               SLOT(4), SLOT(5));
    k_finalize<<<1, 128, 0, stream>>>(SLOT(4), SLOT(5), fg_bn1_g, fg_bn1_b,
                                      SLOT(16), SLOT(17), C_, (float)BNK_);

    k_attn1_stats<<<BN_ / 8, 256, 0, stream>>>(Qb, Kb, knn, xyz, fd_w1, fd_b1,
                                               SLOT(14), SLOT(15), fd_w2, fd_b2,
                                               SLOT(16), SLOT(17), wh + WH_G1, fg_b1,
                                               SLOT(6), SLOT(7));
    k_finalize<<<1, 128, 0, stream>>>(SLOT(6), SLOT(7), fg_bn2_g, fg_bn2_b,
                                      SLOT(18), SLOT(19), ATTN_, (float)BNK_);

    k_main<<<BN_ / 8, 256, 0, stream>>>(Qb, Kb, Vb, knn, xyz, fd_w1, fd_b1,
                                        SLOT(14), SLOT(15), fd_w2, fd_b2,
                                        SLOT(16), SLOT(17), wh + WH_G1, fg_b1,
                                        SLOT(18), SLOT(19), wh + WH_G2, fg_b2,
                                        X1 /*RES*/, SLOT(8), SLOT(9));
    k_finalize<<<1, 128, 0, stream>>>(SLOT(8), SLOT(9), bn2_g, bn2_b,
                                      SLOT(20), SLOT(21), C_, (float)BN_);

    k_gemm_bn<<<BN_ / 16, 256, 0, stream>>>(X1 /*RES*/, wh + WH_W3, SLOT(20), SLOT(21), 1,
                                            Qb /*X3*/, SLOT(10), SLOT(11));
    k_finalize<<<1, 128, 0, stream>>>(SLOT(10), SLOT(11), bn3_g, bn3_b,
                                      SLOT(22), SLOT(23), C_, (float)BN_);

    k_out<<<(int)((BIG_ / 4 + 255) / 256), 256, 0, stream>>>(Qb /*X3*/, SLOT(22), SLOT(23),
                                                             feat, xyz, (float*)d_out);
#undef SLOT
}
